// NetA_NodeOnly_16355235463253
// MI455X (gfx1250) — compile-verified, matrix operand order checked by reading, not run on hardware
//
#include <hip/hip_runtime.h>
#include <hip/hip_bf16.h>
#include <math.h>

// ---------------- problem constants (match reference) ----------------
#define NN    50000     // nodes
#define NE    300000    // edges
#define NG    64        // graphs
#define EMB   64
#define FEAT  8
#define DIN   512       // EMB*FEAT
#define DH    256
#define EDIM  6

typedef _Float16 half8 __attribute__((ext_vector_type(8)));
typedef _Float16 v16h  __attribute__((ext_vector_type(16)));
typedef float    v8f   __attribute__((ext_vector_type(8)));

// ---- ordered-int encoding of float for atomic max -------------------
__device__ __forceinline__ int f2ord(float f) {
    int i = __float_as_int(f);
    return i >= 0 ? i : (i ^ 0x7fffffff);
}
__device__ __forceinline__ float ord2f(int i) {
    return __int_as_float(i >= 0 ? i : (i ^ 0x7fffffff));
}

__device__ __forceinline__ v16h pack16(half8 lo, half8 hi) {
    v16h r;
#pragma unroll
    for (int i = 0; i < 8; ++i) { r[i] = lo[i]; r[i + 8] = hi[i]; }
    return r;
}

// ---------------- small utility kernels ------------------------------
__global__ void k_f32_to_f16(const float* __restrict__ in, _Float16* __restrict__ out, int n) {
    int i = blockIdx.x * blockDim.x + threadIdx.x;
    if (i < n) out[i] = (_Float16)in[i];
}

__global__ void k_embed_gather(const int* __restrict__ idx, const float* __restrict__ emb,
                               _Float16* __restrict__ xh) {
    int i = blockIdx.x * blockDim.x + threadIdx.x;
    if (i >= NN * DIN) return;
    int node = i / DIN, d = i % DIN;
    int f = d / EMB, j = d % EMB;
    int v = idx[node * FEAT + f];
    xh[i] = (_Float16)emb[v * EMB + j];
}

// ---------------- WMMA GEMM:  C[M,Nc] = A[M,K] * W[Nc,K]^T + bias ----
// one wave -> 16 rows x 64 cols (4 accumulators reuse one A fragment)
__global__ void k_gemm_wmma(const _Float16* __restrict__ A,
                            const _Float16* __restrict__ W,
                            const float* __restrict__ bias,
                            float* __restrict__ C,
                            int M, int Kd, int Nc) {
    const int lane = threadIdx.x & 31;
    const int wave = threadIdx.x >> 5;
    const int nb   = Nc >> 6;                        // 64-col groups
    const int mtiles = M >> 4;
    int tile = blockIdx.x * (blockDim.x >> 5) + wave;
    if (tile >= mtiles * nb) return;                 // wave-uniform exit
    const int mt = tile / nb;
    const int ng = tile % nb;

    const int row = mt * 16 + (lane & 15);
    const int kh  = (lane >> 4) * 8;                 // K-half select per ISA layout
    const _Float16* arow = A + (size_t)row * Kd;
    const int c0 = ng * 64 + (lane & 15);
    const _Float16* w0 = W + (size_t)(c0 +  0) * Kd;
    const _Float16* w1 = W + (size_t)(c0 + 16) * Kd;
    const _Float16* w2 = W + (size_t)(c0 + 32) * Kd;
    const _Float16* w3 = W + (size_t)(c0 + 48) * Kd;

    v8f acc0 = {}, acc1 = {}, acc2 = {}, acc3 = {};
    for (int k0 = 0; k0 < Kd; k0 += 32) {
        v16h a = pack16(*(const half8*)(arow + k0 + kh),
                        *(const half8*)(arow + k0 + 16 + kh));
        v16h b0 = pack16(*(const half8*)(w0 + k0 + kh), *(const half8*)(w0 + k0 + 16 + kh));
        v16h b1 = pack16(*(const half8*)(w1 + k0 + kh), *(const half8*)(w1 + k0 + 16 + kh));
        v16h b2 = pack16(*(const half8*)(w2 + k0 + kh), *(const half8*)(w2 + k0 + 16 + kh));
        v16h b3 = pack16(*(const half8*)(w3 + k0 + kh), *(const half8*)(w3 + k0 + 16 + kh));
        acc0 = __builtin_amdgcn_wmma_f32_16x16x32_f16(false, a, false, b0, (short)0, acc0, false, false);
        acc1 = __builtin_amdgcn_wmma_f32_16x16x32_f16(false, a, false, b1, (short)0, acc1, false, false);
        acc2 = __builtin_amdgcn_wmma_f32_16x16x32_f16(false, a, false, b2, (short)0, acc2, false, false);
        acc3 = __builtin_amdgcn_wmma_f32_16x16x32_f16(false, a, false, b3, (short)0, acc3, false, false);
    }
    const float bb0 = bias[c0], bb1 = bias[c0 + 16], bb2 = bias[c0 + 32], bb3 = bias[c0 + 48];
#pragma unroll
    for (int t = 0; t < 8; ++t) {
        int r = mt * 16 + t + ((lane >> 4) << 3);
        float* crow = C + (size_t)r * Nc;
        crow[c0 +  0] = acc0[t] + bb0;
        crow[c0 + 16] = acc1[t] + bb1;
        crow[c0 + 32] = acc2[t] + bb2;
        crow[c0 + 48] = acc3[t] + bb3;
    }
}

// ---------------- edge phase -----------------------------------------
// one wave per edge; lane handles 8 channels; edge projection recomputed
__global__ void k_edge_logits(const int* __restrict__ src, const int* __restrict__ dst,
                              const float* __restrict__ ea,
                              const float* __restrict__ q, const float* __restrict__ k,
                              const float* __restrict__ we, const float* __restrict__ be,
                              float* __restrict__ logits, int* __restrict__ mi) {
    int e = (blockIdx.x * blockDim.x + threadIdx.x) >> 5;
    int lane = threadIdx.x & 31;
    if (e >= NE) return;
    int s = src[e], d = dst[e];
    float a0 = ea[e*EDIM+0], a1 = ea[e*EDIM+1], a2 = ea[e*EDIM+2];
    float a3 = ea[e*EDIM+3], a4 = ea[e*EDIM+4], a5 = ea[e*EDIM+5];
    float acc = 0.f;
    int c0 = lane * 8;
#pragma unroll
    for (int t = 0; t < 8; ++t) {
        int c = c0 + t;
        const float* wr = we + c * EDIM;
        float ev = be[c] + wr[0]*a0 + wr[1]*a1 + wr[2]*a2 + wr[3]*a3 + wr[4]*a4 + wr[5]*a5;
        acc += q[(size_t)d * DH + c] * (k[(size_t)s * DH + c] + ev);
    }
#pragma unroll
    for (int off = 16; off > 0; off >>= 1) acc += __shfl_xor(acc, off, 32);
    if (lane == 0) {
        float lg = acc * 0.0625f;                    // 1/sqrt(256)
        logits[e] = lg;
        atomicMax(&mi[d], f2ord(lg));
    }
}

__global__ void k_m_fix(const int* __restrict__ mi, float* __restrict__ mf, float* __restrict__ s) {
    int n = blockIdx.x * blockDim.x + threadIdx.x;
    if (n >= NN) return;
    float m = ord2f(mi[n]);
    mf[n] = isfinite(m) ? m : 0.0f;
    s[n] = 0.0f;
}

__global__ void k_edge_exp(const int* __restrict__ dst, const float* __restrict__ logits,
                           const float* __restrict__ mf, float* __restrict__ aexp,
                           float* __restrict__ s) {
    int e = blockIdx.x * blockDim.x + threadIdx.x;
    if (e >= NE) return;
    int d = dst[e];
    float a = expf(logits[e] - mf[d]);
    aexp[e] = a;
    atomicAdd(&s[d], a);
}

__global__ void k_edge_agg(const int* __restrict__ src, const int* __restrict__ dst,
                           const float* __restrict__ ea,
                           const float* __restrict__ v,
                           const float* __restrict__ we, const float* __restrict__ be,
                           const float* __restrict__ aexp, const float* __restrict__ s,
                           float* __restrict__ H) {
    int e = (blockIdx.x * blockDim.x + threadIdx.x) >> 5;
    int lane = threadIdx.x & 31;
    if (e >= NE) return;
    int sn = src[e], d = dst[e];
    float coef = aexp[e] / s[d];
    float a0 = ea[e*EDIM+0], a1 = ea[e*EDIM+1], a2 = ea[e*EDIM+2];
    float a3 = ea[e*EDIM+3], a4 = ea[e*EDIM+4], a5 = ea[e*EDIM+5];
    int c0 = lane * 8;
#pragma unroll
    for (int t = 0; t < 8; ++t) {
        int c = c0 + t;
        const float* wr = we + c * EDIM;
        float ev = be[c] + wr[0]*a0 + wr[1]*a1 + wr[2]*a2 + wr[3]*a3 + wr[4]*a4 + wr[5]*a5;
        atomicAdd(&H[(size_t)d * DH + c], (v[(size_t)sn * DH + c] + ev) * coef);
    }
}

__global__ void k_relu_conv(float* __restrict__ H, _Float16* __restrict__ HH) {
    int i = blockIdx.x * blockDim.x + threadIdx.x;
    if (i >= NN * DH) return;
    float x = H[i];
    x = x > 0.f ? x : 0.f;
    H[i] = x;
    HH[i] = (_Float16)x;
}

__global__ void k_init_mi(int* __restrict__ mi) {
    int n = blockIdx.x * blockDim.x + threadIdx.x;
    if (n < NN) mi[n] = 0x807fffff;                  // f2ord(-inf)
}

// ---------------- pooling + MLP --------------------------------------
__global__ void k_pool_init(float* __restrict__ gap, int* __restrict__ gmpi, float* __restrict__ cnt) {
    int i = blockIdx.x * blockDim.x + threadIdx.x;
    if (i >= NG * DH) return;
    gap[i] = 0.f;
    gmpi[i] = 0x807fffff;
    if (i < NG) cnt[i] = 0.f;
}

__global__ void k_pool_acc(const float* __restrict__ h, const int* __restrict__ batch,
                           float* __restrict__ gap, int* __restrict__ gmpi, float* __restrict__ cnt) {
    int i = blockIdx.x * blockDim.x + threadIdx.x;
    if (i >= NN * DH) return;
    int n = i / DH, c = i % DH;
    int b = batch[n];
    float x = h[i];
    atomicAdd(&gap[b * DH + c], x);
    atomicMax(&gmpi[b * DH + c], f2ord(x));
    if (c == 0) atomicAdd(&cnt[b], 1.0f);
}

__global__ void k_pool_mlp(const float* __restrict__ gap, const int* __restrict__ gmpi,
                           const float* __restrict__ cnt,
                           const float* __restrict__ w1, const float* __restrict__ b1,
                           const float* __restrict__ w2, const float* __restrict__ b2,
                           const float* __restrict__ w3, const float* __restrict__ b3,
                           float* __restrict__ out) {
    __shared__ float r[2 * DH];
    __shared__ float o1[256];
    __shared__ float o2[128];
    int g = blockIdx.x, t = threadIdx.x;
    float c = cnt[g]; c = c > 1.f ? c : 1.f;
    r[t] = gap[g * DH + t] / c;
    float m = ord2f(gmpi[g * DH + t]);
    r[DH + t] = isfinite(m) ? m : 0.f;
    __syncthreads();
    float a1 = b1[t];
    for (int j = 0; j < 2 * DH; ++j) a1 += w1[t * 2 * DH + j] * r[j];
    o1[t] = a1 > 0.f ? a1 : 0.f;
    __syncthreads();
    if (t < 128) {
        float a2 = b2[t];
        for (int j = 0; j < 256; ++j) a2 += w2[t * 256 + j] * o1[j];
        o2[t] = a2 > 0.f ? a2 : 0.f;
    }
    __syncthreads();
    if (t == 0) {
        float a3 = b3[0];
        for (int j = 0; j < 128; ++j) a3 += w3[j] * o2[j];
        out[g] = 1.f / (1.f + expf(-a3));
    }
}

// ---------------- host launcher --------------------------------------
extern "C" void kernel_launch(void* const* d_in, const int* in_sizes, int n_in,
                              void* d_out, int out_size, void* d_ws, size_t ws_size,
                              hipStream_t stream) {
    const int*   x_idx  = (const int*)d_in[0];
    const int*   eidx   = (const int*)d_in[1];
    const float* eattr  = (const float*)d_in[2];
    const int*   batch  = (const int*)d_in[3];
    const float* emb    = (const float*)d_in[4];
    const float* c1_wq = (const float*)d_in[5],  *c1_bq = (const float*)d_in[6];
    const float* c1_wk = (const float*)d_in[7],  *c1_bk = (const float*)d_in[8];
    const float* c1_wv = (const float*)d_in[9],  *c1_bv = (const float*)d_in[10];
    const float* c1_we = (const float*)d_in[11], *c1_be = (const float*)d_in[12];
    const float* c1_ws = (const float*)d_in[13], *c1_bs = (const float*)d_in[14];
    const float* c2_wq = (const float*)d_in[15], *c2_bq = (const float*)d_in[16];
    const float* c2_wk = (const float*)d_in[17], *c2_bk = (const float*)d_in[18];
    const float* c2_wv = (const float*)d_in[19], *c2_bv = (const float*)d_in[20];
    const float* c2_we = (const float*)d_in[21], *c2_be = (const float*)d_in[22];
    const float* c2_ws = (const float*)d_in[23], *c2_bs = (const float*)d_in[24];
    const float* w1 = (const float*)d_in[25], *b1 = (const float*)d_in[26];
    const float* w2 = (const float*)d_in[27], *b2 = (const float*)d_in[28];
    const float* w3 = (const float*)d_in[29], *b3 = (const float*)d_in[30];
    float* out = (float*)d_out;

    const int* src = eidx;            // edge_index[0]
    const int* dst = eidx + NE;       // edge_index[1]

    // ---- workspace layout (256B aligned) ----
    char* base = (char*)d_ws;
    size_t off = 0;
    auto alloc = [&](size_t bytes) -> char* {
        char* p = base + off;
        off = (off + bytes + 255) & ~(size_t)255;
        return p;
    };
    _Float16* XH  = (_Float16*)alloc((size_t)NN * DIN * 2);
    float*    Q   = (float*)alloc((size_t)NN * DH * 4);
    float*    K   = (float*)alloc((size_t)NN * DH * 4);
    float*    V   = (float*)alloc((size_t)NN * DH * 4);
    float*    H   = (float*)alloc((size_t)NN * DH * 4);
    _Float16* HH  = (_Float16*)alloc((size_t)NN * DH * 2);
    float*    LOG = (float*)alloc((size_t)NE * 4);
    float*    AEX = (float*)alloc((size_t)NE * 4);
    int*      MI  = (int*)alloc((size_t)NN * 4);
    float*    MF  = (float*)alloc((size_t)NN * 4);
    float*    S   = (float*)alloc((size_t)NN * 4);
    _Float16* WQ1 = (_Float16*)alloc((size_t)DH * DIN * 2);
    _Float16* WK1 = (_Float16*)alloc((size_t)DH * DIN * 2);
    _Float16* WV1 = (_Float16*)alloc((size_t)DH * DIN * 2);
    _Float16* WS1 = (_Float16*)alloc((size_t)DH * DIN * 2);
    _Float16* WQ2 = (_Float16*)alloc((size_t)DH * DH * 2);
    _Float16* WK2 = (_Float16*)alloc((size_t)DH * DH * 2);
    _Float16* WV2 = (_Float16*)alloc((size_t)DH * DH * 2);
    _Float16* WS2 = (_Float16*)alloc((size_t)DH * DH * 2);
    float*    GAP = (float*)alloc((size_t)NG * DH * 4);
    int*      GMP = (int*)alloc((size_t)NG * DH * 4);
    float*    CNT = (float*)alloc((size_t)NG * 4);

    const int T = 256;
    // ---- weight conversion to f16 ----
    int n1 = DH * DIN, n2 = DH * DH;
    k_f32_to_f16<<<(n1 + T - 1) / T, T, 0, stream>>>(c1_wq, WQ1, n1);
    k_f32_to_f16<<<(n1 + T - 1) / T, T, 0, stream>>>(c1_wk, WK1, n1);
    k_f32_to_f16<<<(n1 + T - 1) / T, T, 0, stream>>>(c1_wv, WV1, n1);
    k_f32_to_f16<<<(n1 + T - 1) / T, T, 0, stream>>>(c1_ws, WS1, n1);
    k_f32_to_f16<<<(n2 + T - 1) / T, T, 0, stream>>>(c2_wq, WQ2, n2);
    k_f32_to_f16<<<(n2 + T - 1) / T, T, 0, stream>>>(c2_wk, WK2, n2);
    k_f32_to_f16<<<(n2 + T - 1) / T, T, 0, stream>>>(c2_wv, WV2, n2);
    k_f32_to_f16<<<(n2 + T - 1) / T, T, 0, stream>>>(c2_ws, WS2, n2);

    // ---- embedding gather -> f16 activations ----
    k_embed_gather<<<((size_t)NN * DIN + T - 1) / T, T, 0, stream>>>(x_idx, emb, XH);

    const int gemm1_blocks = ((NN / 16) * (DH / 64) + 7) / 8;   // 8 waves/block
    const int gemm2_blocks = gemm1_blocks;
    const int edge_wave_blocks = (NE + 7) / 8;
    const int edge_thr_blocks  = (NE + T - 1) / T;
    const int node_blocks = (NN + T - 1) / T;
    const int nd_blocks = ((size_t)NN * DH + T - 1) / T;

    // ================= layer 1 =================
    k_gemm_wmma<<<gemm1_blocks, T, 0, stream>>>(XH, WQ1, c1_bq, Q, NN, DIN, DH);
    k_gemm_wmma<<<gemm1_blocks, T, 0, stream>>>(XH, WK1, c1_bk, K, NN, DIN, DH);
    k_gemm_wmma<<<gemm1_blocks, T, 0, stream>>>(XH, WV1, c1_bv, V, NN, DIN, DH);
    k_gemm_wmma<<<gemm1_blocks, T, 0, stream>>>(XH, WS1, c1_bs, H, NN, DIN, DH); // skip init
    k_init_mi<<<node_blocks, T, 0, stream>>>(MI);
    k_edge_logits<<<edge_wave_blocks, T, 0, stream>>>(src, dst, eattr, Q, K, c1_we, c1_be, LOG, MI);
    k_m_fix<<<node_blocks, T, 0, stream>>>(MI, MF, S);
    k_edge_exp<<<edge_thr_blocks, T, 0, stream>>>(dst, LOG, MF, AEX, S);
    k_edge_agg<<<edge_wave_blocks, T, 0, stream>>>(src, dst, eattr, V, c1_we, c1_be, AEX, S, H);
    k_relu_conv<<<nd_blocks, T, 0, stream>>>(H, HH);

    // ================= layer 2 =================
    k_gemm_wmma<<<gemm2_blocks, T, 0, stream>>>(HH, WQ2, c2_bq, Q, NN, DH, DH);
    k_gemm_wmma<<<gemm2_blocks, T, 0, stream>>>(HH, WK2, c2_bk, K, NN, DH, DH);
    k_gemm_wmma<<<gemm2_blocks, T, 0, stream>>>(HH, WV2, c2_bv, V, NN, DH, DH);
    k_gemm_wmma<<<gemm2_blocks, T, 0, stream>>>(HH, WS2, c2_bs, H, NN, DH, DH); // overwrites: skip init
    k_init_mi<<<node_blocks, T, 0, stream>>>(MI);
    k_edge_logits<<<edge_wave_blocks, T, 0, stream>>>(src, dst, eattr, Q, K, c2_we, c2_be, LOG, MI);
    k_m_fix<<<node_blocks, T, 0, stream>>>(MI, MF, S);
    k_edge_exp<<<edge_thr_blocks, T, 0, stream>>>(dst, LOG, MF, AEX, S);
    k_edge_agg<<<edge_wave_blocks, T, 0, stream>>>(src, dst, eattr, V, c2_we, c2_be, AEX, S, H);
    k_relu_conv<<<nd_blocks, T, 0, stream>>>(H, HH);

    // ================= pooling + MLP =================
    k_pool_init<<<(NG * DH + T - 1) / T, T, 0, stream>>>(GAP, GMP, CNT);
    k_pool_acc<<<nd_blocks, T, 0, stream>>>(H, batch, GAP, GMP, CNT);
    k_pool_mlp<<<NG, 256, 0, stream>>>(GAP, GMP, CNT, w1, b1, w2, b2, w3, b3, out);
}
